// EchoLayer_69836168233464
// MI455X (gfx1250) — matrix operand order repaired and checked by reading, not compiled, not run on hardware
//
#include <hip/hip_runtime.h>
#include <cstdint>
#include <cstddef>

#define B_   32
#define S_   1024
#define H_   512
#define D_   3
#define NBLK 32                         // one WG per 16-column chunk of H
#define NTHR 256                        // 8 waves (wave32)
#define RS_  (H_ + 8)                   // LDS weight row stride: 260 dwords -> banks 4n mod 64
#define ACC_FLOATS (6 * B_ * 16)        // 6 planes (r,z,n_ih,n_hh,outG,outT) of 32x16 f32
#define WROWS (D_ * 2 * 3 * 16)         // 288 weight rows (lvl x {ih,hh} x gate x 16 cols)
#define WLDS_BYTES (WROWS * RS_ * 2)    // 299,520
#define HLDS_FLOATS (D_ * B_ * 16)      // per-WG h state slice, 6,144 B
#define BIAS_FLOATS ((D_ * 4 + 2) * 16) // rsum,zsum,ni,nh per lvl + bg,bt
#define LDS_BYTES (ACC_FLOATS * 4 + WLDS_BYTES + HLDS_FLOATS * 4 + 1024)  // 318,976 <= 320KB

typedef __attribute__((ext_vector_type(16))) __bf16 v16bf;
typedef __attribute__((ext_vector_type(8)))  __bf16 v8bf;
typedef __attribute__((ext_vector_type(8)))  float  v8f;

__device__ __forceinline__ v8f wmma_bf16(v16bf a, v16bf b, v8f c) {
  return __builtin_amdgcn_wmma_f32_16x16x32_bf16(
      false, a, false, b, (short)0, c, false, false);
}

// A tile (row-major [M,K], bf16): lanes 0-15 hold K=0..15 of row M=lane,
// lanes 16-31 hold K=16..31.
__device__ __forceinline__ v16bf load_a(const __bf16* base, size_t ld,
                                        int mtile, int k0, int lane) {
  const int m  = mtile * 16 + (lane & 15);
  const int kb = (lane >> 4) * 16;
  const __bf16* p = base + (size_t)m * ld + (size_t)(k0 + kb);
  v8bf c0 = *(const v8bf*)(p);
  v8bf c1 = *(const v8bf*)(p + 8);
  v16bf a;
#pragma unroll
  for (int i = 0; i < 8; ++i) { a[i] = c0[i]; a[i + 8] = c1[i]; }
  return a;
}

// B tile from global: B(k,n) = W[n,k], W row-major [N,K]; lane holds 16 contiguous k.
__device__ __forceinline__ v16bf load_b(const __bf16* w, size_t ld,
                                        int n0, int k0, int lane) {
  const int n  = n0 + (lane & 15);
  const int kb = (lane >> 4) * 16;
  return *(const v16bf*)(w + (size_t)n * ld + (size_t)(k0 + kb));
}

// B tile from LDS weight cache (row stride RS_), 16B accesses -> ds_load_b128.
__device__ __forceinline__ v16bf load_b_lds(const __bf16* plane, int k0, int lane) {
  const __bf16* p = plane + (size_t)(lane & 15) * RS_ + (size_t)(k0 + ((lane >> 4) * 16));
  v8bf c0 = *(const v8bf*)(p);
  v8bf c1 = *(const v8bf*)(p + 8);
  v16bf b;
#pragma unroll
  for (int i = 0; i < 8; ++i) { b[i] = c0[i]; b[i + 8] = c1[i]; }
  return b;
}

// Monotonic-counter grid barrier; counter zeroed each launch by zero_ws.
__device__ __forceinline__ void grid_sync(unsigned* cnt, unsigned target) {
  __syncthreads();
  if (threadIdx.x == 0) {
    __threadfence();
    atomicAdd(cnt, 1u);
    while (__hip_atomic_load(cnt, __ATOMIC_ACQUIRE, __HIP_MEMORY_SCOPE_AGENT) < target)
      __builtin_amdgcn_s_sleep(1);
    __threadfence();
  }
  __syncthreads();
}

__device__ __forceinline__ float sigmoidf_(float v) {
  return 1.0f / (1.0f + __expf(-v));
}
__device__ __forceinline__ float tanhf_(float v) {
  float e = __expf(2.0f * v);
  return 1.0f - 2.0f / (e + 1.0f);
}

__global__ void cvt_bf16(const float* __restrict__ s, __bf16* __restrict__ d, int n) {
  for (int i = blockIdx.x * blockDim.x + threadIdx.x; i < n;
       i += gridDim.x * blockDim.x)
    d[i] = (__bf16)s[i];
}

__global__ void zero_ws(uint32_t* p, int nwords) {
  for (int i = blockIdx.x * blockDim.x + threadIdx.x; i < nwords;
       i += gridDim.x * blockDim.x)
    p[i] = 0u;
}

__global__ __launch_bounds__(NTHR, 1)
void echo_persistent(const float* __restrict__ x,
                     const float* __restrict__ bih, const float* __restrict__ bhh,
                     const float* __restrict__ bt,  const float* __restrict__ bg,
                     const __bf16* __restrict__ xBf,
                     const __bf16* __restrict__ wihBf, const __bf16* __restrict__ whhBf,
                     const __bf16* __restrict__ wtBf,  const __bf16* __restrict__ wgBf,
                     __bf16* hBf,                       // [2 parity][D][B][H] bf16 mirrors
                     __bf16* curA, __bf16* curB, __bf16* combBf,
                     unsigned* cnt, float* out) {
  extern __shared__ char smem_raw[];
  float*  acc_s   = (float*)smem_raw;
  __bf16* wlds    = (__bf16*)(smem_raw + ACC_FLOATS * 4);
  float*  hLds    = (float*)(smem_raw + ACC_FLOATS * 4 + WLDS_BYTES);
  float*  biasLds = (float*)(smem_raw + ACC_FLOATS * 4 + WLDS_BYTES + HLDS_FLOATS * 4);
#define ACC(g, m, j)  acc_s[(((g) * B_) + (m)) * 16 + (j)]
#define HLDS(l, m, j) hLds[(((l) * B_) + (m)) * 16 + (j)]

  const int lane = threadIdx.x & 31;
  const int w    = threadIdx.x >> 5;
  const int col0 = blockIdx.x * 16;
  const unsigned P = gridDim.x;
  unsigned phase = 0;

  // ---- One-time LDS setup ------------------------------------------------
  for (int idx = threadIdx.x; idx < WROWS * 64; idx += NTHR) {
    const int r    = idx >> 6;
    const int c    = idx & 63;
    const int lvl  = r / 96;
    const int rem  = r % 96;
    const int mat  = rem / 48;          // 0 = wih, 1 = whh
    const int rem2 = rem % 48;
    const int gate = rem2 / 16;
    const int nn   = rem2 % 16;
    const __bf16* g = (mat ? whhBf : wihBf) + (size_t)lvl * 3 * H_ * H_ +
                      (size_t)(gate * H_ + col0 + nn) * H_;
    *(uint4*)(wlds + (size_t)r * RS_ + c * 8) = *(const uint4*)(g + c * 8);
  }
  for (int i = threadIdx.x; i < HLDS_FLOATS; i += NTHR) hLds[i] = 0.0f;
  for (int i = threadIdx.x; i < BIAS_FLOATS; i += NTHR) {
    const int row = i >> 4;
    const int jj  = i & 15;
    const int j   = col0 + jj;
    float v;
    if (row < D_ * 4) {
      const int lvl  = row >> 2;
      const int kind = row & 3;         // 0=rsum 1=zsum 2=ni 3=nh
      const int bo   = lvl * 3 * H_;
      if (kind == 0)      v = bih[bo + j] + bhh[bo + j];
      else if (kind == 1) v = bih[bo + H_ + j] + bhh[bo + H_ + j];
      else if (kind == 2) v = bih[bo + 2 * H_ + j];
      else                v = bhh[bo + 2 * H_ + j];
    } else {
      v = (row == D_ * 4) ? bg[j] : bt[j];
    }
    biasLds[i] = v;
  }
  __syncthreads();

  for (int t = 0; t < S_; ++t) {
    const int par = t & 1;
    const __bf16* hRead  = hBf + (size_t)par       * (D_ * B_ * H_);
    __bf16*       hWrite = hBf + (size_t)(1 - par) * (D_ * B_ * H_);

    for (int lvl = 0; lvl < D_; ++lvl) {
      if (w < 3) {                              // GRU GEMM: wave = gate, both mtiles
        const int gate = w;                     // 0=r 1=z 2=n
        const __bf16* Aih = (lvl == 0) ? (xBf + (size_t)t * H_)
                                       : (lvl == 1 ? curA : curB);
        const size_t  ldA = (lvl == 0) ? (size_t)S_ * H_ : (size_t)H_;
        const __bf16* Ahh = hRead + (size_t)lvl * B_ * H_;
        const __bf16* Pih = wlds + (size_t)(((lvl * 2 + 0) * 3 + gate) * 16) * RS_;
        const __bf16* Phh = wlds + (size_t)(((lvl * 2 + 1) * 3 + gate) * 16) * RS_;
        v8f accI0 = {}, accI1 = {}, accH0 = {}, accH1 = {};
#pragma unroll 2
        for (int k0 = 0; k0 < H_; k0 += 32) {
          v16bf bI  = load_b_lds(Pih, k0, lane);     // shared across mtiles
          v16bf bH  = load_b_lds(Phh, k0, lane);
          v16bf aI0 = load_a(Aih, ldA, 0, k0, lane);
          v16bf aI1 = load_a(Aih, ldA, 1, k0, lane);
          v16bf aH0 = load_a(Ahh, H_,  0, k0, lane);
          v16bf aH1 = load_a(Ahh, H_,  1, k0, lane);
          accI0 = wmma_bf16(aI0, bI, accI0);         // 4 independent chains:
          accI1 = wmma_bf16(aI1, bI, accI1);         // no D->A/B hazards
          accH0 = wmma_bf16(aH0, bH, accH0);
          accH1 = wmma_bf16(aH1, bH, accH1);
        }
#pragma unroll
        for (int e = 0; e < 8; ++e) {
          const int mr = e + 8 * (lane >> 4);
          const int jj = lane & 15;
          if (gate < 2) {
            ACC(gate, mr,      jj) = accI0[e] + accH0[e];
            ACC(gate, mr + 16, jj) = accI1[e] + accH1[e];
          } else {
            ACC(2, mr,      jj) = accI0[e];
            ACC(2, mr + 16, jj) = accI1[e];
            ACC(3, mr,      jj) = accH0[e];
            ACC(3, mr + 16, jj) = accH1[e];
          }
        }
      } else if (lvl == 0 && t > 0 && w >= 4 && w < 6) {
        // Fused output stage for step t-1 (waves 4,5): one matrix each,
        // both mtiles, B loaded once per k-step.
        const int mat = w - 4;                  // 0 = wg (gate), 1 = wt
        const __bf16* W = mat ? wtBf : wgBf;    // [H, 2H]
        const int to = t - 1;
        v8f acc0 = {}, acc1 = {};
#pragma unroll 4
        for (int k0 = 0; k0 < 2 * H_; k0 += 32) {
          const __bf16* Asrc = (k0 < H_) ? (xBf + (size_t)to * H_) : combBf;
          const size_t  ldAs = (k0 < H_) ? (size_t)S_ * H_ : (size_t)H_;
          const int     ks   = (k0 < H_) ? k0 : (k0 - H_);
          v16bf bm = load_b(W, 2 * H_, col0, k0, lane);
          v16bf a0 = load_a(Asrc, ldAs, 0, ks, lane);
          v16bf a1 = load_a(Asrc, ldAs, 1, ks, lane);
          acc0 = wmma_bf16(a0, bm, acc0);
          acc1 = wmma_bf16(a1, bm, acc1);
        }
#pragma unroll
        for (int e = 0; e < 8; ++e) {
          const int mr = e + 8 * (lane >> 4);
          ACC(4 + mat, mr,      lane & 15) = acc0[e];
          ACC(4 + mat, mr + 16, lane & 15) = acc1[e];
        }
      }
      __syncthreads();

      // GRU elementwise for this WG's 32x16 slice (LDS-only inputs)
      for (int i = threadIdx.x; i < B_ * 16; i += NTHR) {
        const int b  = i >> 4;
        const int jj = i & 15;
        const int j  = col0 + jj;
        const float* bl = biasLds + (lvl * 4) * 16;
        float r  = sigmoidf_(ACC(0, b, jj) + bl[jj]);
        float z  = sigmoidf_(ACC(1, b, jj) + bl[16 + jj]);
        float n  = tanhf_(ACC(2, b, jj) + bl[32 + jj] +
                          r * (ACC(3, b, jj) + bl[48 + jj]));
        float hp = HLDS(lvl, b, jj);
        float hn = (1.0f - z) * n + z * hp;
        HLDS(lvl, b, jj) = hn;
        hWrite[(size_t)lvl * B_ * H_ + (size_t)b * H_ + j] = (__bf16)hn;
        if (lvl == 0)      curA[b * H_ + j] = (__bf16)(0.95f * hn);
        else if (lvl == 1) curB[b * H_ + j] = (__bf16)(0.95f * hn);
        else {
          float comb = (HLDS(0, b, jj) + HLDS(1, b, jj) + hn) * (1.0f / 3.0f);
          combBf[b * H_ + j] = (__bf16)comb;
        }
      }
      if (lvl == 0 && t > 0) {                  // output elementwise for t-1
        const int to = t - 1;
        for (int i = threadIdx.x; i < B_ * 16; i += NTHR) {
          const int b  = i >> 4;
          const int jj = i & 15;
          const int j  = col0 + jj;
          float g  = sigmoidf_(ACC(4, b, jj) + biasLds[(D_ * 4) * 16 + jj]);
          float tr = ACC(5, b, jj) + biasLds[(D_ * 4 + 1) * 16 + jj];
          const size_t xo = (size_t)b * S_ * H_ + (size_t)to * H_ + j;
          float xv = x[xo];
          out[xo] = g * tr + (1.0f - g) * xv;
        }
      }
      grid_sync(cnt, (++phase) * P);            // 3 barriers per step
    }
  }

  // ---- Epilogue: output stage for t = S-1 --------------------------------
  if (w >= 4 && w < 6) {
    const int mat = w - 4;
    const __bf16* W = mat ? wtBf : wgBf;
    const int to = S_ - 1;
    v8f acc0 = {}, acc1 = {};
#pragma unroll 4
    for (int k0 = 0; k0 < 2 * H_; k0 += 32) {
      const __bf16* Asrc = (k0 < H_) ? (xBf + (size_t)to * H_) : combBf;
      const size_t  ldAs = (k0 < H_) ? (size_t)S_ * H_ : (size_t)H_;
      const int     ks   = (k0 < H_) ? k0 : (k0 - H_);
      v16bf bm = load_b(W, 2 * H_, col0, k0, lane);
      v16bf a0 = load_a(Asrc, ldAs, 0, ks, lane);
      v16bf a1 = load_a(Asrc, ldAs, 1, ks, lane);
      acc0 = wmma_bf16(a0, bm, acc0);
      acc1 = wmma_bf16(a1, bm, acc1);
    }
#pragma unroll
    for (int e = 0; e < 8; ++e) {
      const int mr = e + 8 * (lane >> 4);
      ACC(4 + mat, mr,      lane & 15) = acc0[e];
      ACC(4 + mat, mr + 16, lane & 15) = acc1[e];
    }
  }
  __syncthreads();
  {
    const int to = S_ - 1;
    for (int i = threadIdx.x; i < B_ * 16; i += NTHR) {
      const int b  = i >> 4;
      const int jj = i & 15;
      const int j  = col0 + jj;
      float g  = sigmoidf_(ACC(4, b, jj) + biasLds[(D_ * 4) * 16 + jj]);
      float tr = ACC(5, b, jj) + biasLds[(D_ * 4 + 1) * 16 + jj];
      const size_t xo = (size_t)b * S_ * H_ + (size_t)to * H_ + j;
      float xv = x[xo];
      out[xo] = g * tr + (1.0f - g) * xv;
    }
  }

  // Final state tail of d_out: [D,B,H] fp32, straight from this WG's LDS slice
  for (int i = threadIdx.x; i < D_ * B_ * 16; i += NTHR) {
    const int lvl = i / (B_ * 16);
    const int rem = i % (B_ * 16);
    const int b   = rem >> 4;
    const int jj  = rem & 15;
    out[(size_t)B_ * S_ * H_ + (size_t)lvl * B_ * H_ + (size_t)b * H_ + col0 + jj] =
        HLDS(lvl, b, jj);
  }
#undef ACC
#undef HLDS
}

extern "C" void kernel_launch(void* const* d_in, const int* in_sizes, int n_in,
                              void* d_out, int out_size, void* d_ws, size_t ws_size,
                              hipStream_t stream) {
  const float* x   = (const float*)d_in[0];
  const float* wih = (const float*)d_in[1];
  const float* whh = (const float*)d_in[2];
  const float* bih = (const float*)d_in[3];
  const float* bhh = (const float*)d_in[4];
  const float* wt  = (const float*)d_in[5];
  const float* bt  = (const float*)d_in[6];
  const float* wg  = (const float*)d_in[7];
  const float* bg  = (const float*)d_in[8];
  float* out = (float*)d_out;

  char* ws = (char*)d_ws;
  size_t off = 0;
  auto carve = [&](size_t bytes) -> char* {
    char* p = ws + off;
    off += (bytes + 255) & ~(size_t)255;
    return p;
  };
  const int nX   = B_ * S_ * H_;
  const int nWih = D_ * 3 * H_ * H_;
  const int nWt  = H_ * 2 * H_;

  __bf16* xBf   = (__bf16*)carve((size_t)nX * 2);
  __bf16* wihBf = (__bf16*)carve((size_t)nWih * 2);
  __bf16* whhBf = (__bf16*)carve((size_t)nWih * 2);
  __bf16* wtBf  = (__bf16*)carve((size_t)nWt * 2);
  __bf16* wgBf  = (__bf16*)carve((size_t)nWt * 2);

  char* zero_base = ws + off;
  __bf16* hBf    = (__bf16*)carve((size_t)2 * D_ * B_ * H_ * 2);
  __bf16* curA   = (__bf16*)carve((size_t)B_ * H_ * 2);
  __bf16* curB   = (__bf16*)carve((size_t)B_ * H_ * 2);
  __bf16* combBf = (__bf16*)carve((size_t)B_ * H_ * 2);
  unsigned* cnt  = (unsigned*)carve(128);
  const int zwords = (int)(((ws + off) - zero_base) / 4);

  cvt_bf16<<<2048, 256, 0, stream>>>(x,   xBf,   nX);
  cvt_bf16<<<1024, 256, 0, stream>>>(wih, wihBf, nWih);
  cvt_bf16<<<1024, 256, 0, stream>>>(whh, whhBf, nWih);
  cvt_bf16<<<256,  256, 0, stream>>>(wt,  wtBf,  nWt);
  cvt_bf16<<<256,  256, 0, stream>>>(wg,  wgBf,  nWt);
  zero_ws<<<128, 256, 0, stream>>>((uint32_t*)zero_base, zwords);

  (void)hipFuncSetAttribute(reinterpret_cast<const void*>(echo_persistent),
                            hipFuncAttributeMaxDynamicSharedMemorySize,
                            LDS_BYTES);
  echo_persistent<<<NBLK, NTHR, LDS_BYTES, stream>>>(
      x, bih, bhh, bt, bg,
      xBf, wihBf, whhBf, wtBf, wgBf,
      hBf, curA, curB, combBf, cnt, out);
}
